// ScaledDotAttn_43525198578143
// MI455X (gfx1250) — compile-verified, matrix operand order checked by reading, not run on hardware
//
#include <hip/hip_runtime.h>
#include <hip/hip_bf16.h>

// ---------------------------------------------------------------------------
// ScaledDotAttn for MI455X (gfx1250, wave32, WMMA, async-to-LDS copies)
//   q = input_q @ Wq + bq ; k = input_k @ Wk + bk          (bf16 WMMA GEMM)
//   S = (q k^T)/sqrt(H) + outer(m, m)                      (bf16 WMMA GEMM)
//   P = softmax(S)                                         (fp32 rowwise)
//   out = P @ input_k                                      (bf16 WMMA GEMM)
// GEMM geometry: 128x128 block tile, 4 waves, 64x64 per wave (4x4 WMMA),
// BK=32. Pure bf16 tile copies use global_load_async_to_lds_b128 (ASYNCcnt).
// ---------------------------------------------------------------------------

typedef __attribute__((ext_vector_type(16))) __bf16 v16bf;
typedef __attribute__((ext_vector_type(8)))  float  v8f;

#define B_ 8
#define L_ 2048
#define D_ 1024
#define H_ 1024
#define M_TOTAL (B_ * L_)  // 16384

// Workspace layout (bytes). Total ~225 MB.
#define OFF_V 0ull                                      // bf16 V   33.6 MB
#define OFF_Q (OFF_V + (size_t)M_TOTAL * D_ * 2)        // bf16 Q   33.6 MB
#define OFF_K (OFF_Q + (size_t)M_TOTAL * H_ * 2)        // bf16 K   33.6 MB
#define OFF_M (OFF_K + (size_t)M_TOTAL * H_ * 2)        // f32 mask vals
#define OFF_S (OFF_M + (size_t)M_TOTAL * 4)             // f32 S    134 MB
#define OFF_P OFF_Q  // bf16 P (67.1 MB) aliases Q+K exactly, used after S

// Async copy of 16 bytes global -> LDS, tracked by ASYNCcnt.
__device__ __forceinline__ void async_b128(const void* g, const void* lds) {
  const unsigned long long ga = (unsigned long long)g;
  const unsigned la = (unsigned)(size_t)lds;  // low 32 bits = LDS byte addr
  asm volatile("global_load_async_to_lds_b128 %0, %1, off"
               :: "v"(la), "v"(ga) : "memory");
}
__device__ __forceinline__ void async_wait0() {
  asm volatile("s_wait_asynccnt 0x0" ::: "memory");
}

// ---------------------------------------------------------------------------
// Kernel 0: V = bf16(input_k); mvals = (k_mask > .5) ? 0 : -1e9
// ---------------------------------------------------------------------------
__global__ __launch_bounds__(256) void prep_kernel(
    const float* __restrict__ input_k, const float* __restrict__ k_mask,
    __bf16* __restrict__ Vb, float* __restrict__ mvals) {
  const int idx = blockIdx.x * 256 + threadIdx.x;
  if (idx < M_TOTAL * D_) Vb[idx] = (__bf16)input_k[idx];
  if (idx < M_TOTAL) mvals[idx] = (k_mask[idx] > 0.5f) ? 0.0f : -1.0e9f;
}

// ---------------------------------------------------------------------------
// Kernel 1: C_bf16[M,N] = A_f32[M,K] @ W_f32[K,N] + bias[N]
//   f32 sources need on-the-fly cvt -> VGPR staging path (no async).
// ---------------------------------------------------------------------------
__global__ __launch_bounds__(128) void proj_gemm_kernel(
    const float* __restrict__ A, const float* __restrict__ W,
    const float* __restrict__ bias, __bf16* __restrict__ C,
    int M, int N, int K) {
  __shared__ __align__(64) __bf16 As[128][32];  // [m][k]
  __shared__ __align__(64) __bf16 Bs[128][32];  // transposed: [n][k]

  const int t = threadIdx.x;
  const int lane = t & 31;
  const int wave = t >> 5;        // 0..3
  const int waveM = wave & 1;     // 2 row groups of 64
  const int waveN = wave >> 1;    // 2 col groups of 64
  const int m0 = blockIdx.y * 128;
  const int n0 = blockIdx.x * 128;

  v8f acc[4][4];
#pragma unroll
  for (int i = 0; i < 4; ++i)
#pragma unroll
    for (int j = 0; j < 4; ++j) acc[i][j] = (v8f){0, 0, 0, 0, 0, 0, 0, 0};

  const int bKr = t >> 2;        // 0..31
  const int bNc = (t & 3) * 32;  // 0,32,64,96

  for (int k0 = 0; k0 < K; k0 += 32) {
    // Stage A tile: one 32-elem row per thread, f32 -> bf16
    const float* ag = A + (size_t)(m0 + t) * K + k0;
#pragma unroll
    for (int j = 0; j < 32; ++j) As[t][j] = (__bf16)ag[j];
    // Stage W tile transposed: Bs[n][k]
    const float* wg = W + (size_t)(k0 + bKr) * N + n0 + bNc;
#pragma unroll
    for (int j = 0; j < 32; ++j) Bs[bNc + j][bKr] = (__bf16)wg[j];

    if (k0 + 32 < K) {  // global_prefetch_b8 for next K-step
      __builtin_prefetch(ag + 32, 0, 1);
      __builtin_prefetch(W + (size_t)(k0 + 32 + bKr) * N + n0 + bNc, 0, 1);
    }
    __syncthreads();

    // Fragments in contiguous k order (A and B alike): lanes 0-15 hold K 0..15,
    // lanes 16-31 hold K 16..31 (one 32B chunk per lane)
    v16bf af[4], bf[4];
#pragma unroll
    for (int i = 0; i < 4; ++i)
      af[i] = *(const v16bf*)&As[waveM * 64 + i * 16 + (lane & 15)][(lane >> 4) * 16];
#pragma unroll
    for (int j = 0; j < 4; ++j)
      bf[j] = *(const v16bf*)&Bs[waveN * 64 + j * 16 + (lane & 15)][(lane >> 4) * 16];

#pragma unroll
    for (int i = 0; i < 4; ++i)
#pragma unroll
      for (int j = 0; j < 4; ++j)
        acc[i][j] = __builtin_amdgcn_wmma_f32_16x16x32_bf16(
            false, af[i], false, bf[j], (short)0, acc[i][j], false, false);
    __syncthreads();
  }

  // C/D layout: VGPR e, lanes<16: M=e ; lanes>=16: M=8+e ; N=lane&15
#pragma unroll
  for (int i = 0; i < 4; ++i) {
    const int rBase = m0 + waveM * 64 + i * 16 + ((lane >> 4) ? 8 : 0);
#pragma unroll
    for (int j = 0; j < 4; ++j) {
      const int c = n0 + waveN * 64 + j * 16 + (lane & 15);
      const float bv = bias[c];
#pragma unroll
      for (int e = 0; e < 8; ++e)
        C[(size_t)(rBase + e) * N + c] = (__bf16)(acc[i][j][e] + bv);
    }
  }
}

// ---------------------------------------------------------------------------
// Kernel 2: S[b,q,k] = (Q[b,q,:]·K[b,k,:]) / 32 + m[b,q]*m[b,k]
//   Both tiles are pure bf16 copies -> async-to-LDS path.
// ---------------------------------------------------------------------------
__global__ __launch_bounds__(128) void qk_scores_kernel(
    const __bf16* __restrict__ Qb, const __bf16* __restrict__ Kb,
    const float* __restrict__ mvals, float* __restrict__ S) {
  __shared__ __align__(64) __bf16 Qs[128][32];
  __shared__ __align__(64) __bf16 Ks[128][32];
  __shared__ float rowM[128], colM[128];

  const int t = threadIdx.x;
  const int lane = t & 31;
  const int wave = t >> 5;
  const int waveM = wave & 1, waveN = wave >> 1;
  const int b = blockIdx.z;
  const int q0 = blockIdx.y * 128;
  const int k0 = blockIdx.x * 128;

  rowM[t] = mvals[b * L_ + q0 + t];
  colM[t] = mvals[b * L_ + k0 + t];

  const __bf16* Qbase = Qb + (size_t)b * L_ * H_;
  const __bf16* Kbase = Kb + (size_t)b * L_ * H_;

  v8f acc[4][4];
#pragma unroll
  for (int i = 0; i < 4; ++i)
#pragma unroll
    for (int j = 0; j < 4; ++j) acc[i][j] = (v8f){0, 0, 0, 0, 0, 0, 0, 0};

  for (int h0 = 0; h0 < H_; h0 += 32) {
    // One 64-byte row per thread per tile: 4 async B128 copies each.
    const char* qg = (const char*)&Qbase[(size_t)(q0 + t) * H_ + h0];
    const char* kg = (const char*)&Kbase[(size_t)(k0 + t) * H_ + h0];
    const char* ql = (const char*)&Qs[t][0];
    const char* kl = (const char*)&Ks[t][0];
#pragma unroll
    for (int j = 0; j < 4; ++j) {
      async_b128(qg + j * 16, ql + j * 16);
      async_b128(kg + j * 16, kl + j * 16);
    }
    async_wait0();
    __syncthreads();

    v16bf qf[4], kf[4];
#pragma unroll
    for (int i = 0; i < 4; ++i)
      qf[i] = *(const v16bf*)&Qs[waveM * 64 + i * 16 + (lane & 15)][(lane >> 4) * 16];
#pragma unroll
    for (int j = 0; j < 4; ++j)
      kf[j] = *(const v16bf*)&Ks[waveN * 64 + j * 16 + (lane & 15)][(lane >> 4) * 16];

#pragma unroll
    for (int i = 0; i < 4; ++i)
#pragma unroll
      for (int j = 0; j < 4; ++j)
        acc[i][j] = __builtin_amdgcn_wmma_f32_16x16x32_bf16(
            false, qf[i], false, kf[j], (short)0, acc[i][j], false, false);
    __syncthreads();
  }

  const float scale = 0.03125f;  // 1/sqrt(1024)
  float* Sb = S + (size_t)b * L_ * L_;
#pragma unroll
  for (int i = 0; i < 4; ++i) {
    const int rl = waveM * 64 + i * 16 + ((lane >> 4) ? 8 : 0);
#pragma unroll
    for (int j = 0; j < 4; ++j) {
      const int cl = waveN * 64 + j * 16 + (lane & 15);
      const float mk = colM[cl];
#pragma unroll
      for (int e = 0; e < 8; ++e)
        Sb[(size_t)(q0 + rl + e) * L_ + (k0 + cl)] =
            acc[i][j][e] * scale + rowM[rl + e] * mk;
    }
  }
}

// ---------------------------------------------------------------------------
// Kernel 3: rowwise softmax, S f32 -> P bf16 (one row of 2048 per block)
// ---------------------------------------------------------------------------
__global__ __launch_bounds__(256) void softmax_kernel(
    const float* __restrict__ S, __bf16* __restrict__ P) {
  const int r = blockIdx.x;  // b*L + q
  const int t = threadIdx.x;
  const float* srow = S + (size_t)r * L_;

  float v[8];
#pragma unroll
  for (int j = 0; j < 8; ++j) v[j] = srow[t + j * 256];
  float m = v[0];
#pragma unroll
  for (int j = 1; j < 8; ++j) m = fmaxf(m, v[j]);

  __shared__ float red[256];
  red[t] = m;
  __syncthreads();
  for (int s = 128; s > 0; s >>= 1) {
    if (t < s) red[t] = fmaxf(red[t], red[t + s]);
    __syncthreads();
  }
  m = red[0];
  __syncthreads();

  float e[8], sum = 0.0f;
#pragma unroll
  for (int j = 0; j < 8; ++j) {
    e[j] = __expf(v[j] - m);
    sum += e[j];
  }
  red[t] = sum;
  __syncthreads();
  for (int s = 128; s > 0; s >>= 1) {
    if (t < s) red[t] += red[t + s];
    __syncthreads();
  }
  const float inv = 1.0f / red[0];

  __bf16* prow = P + (size_t)r * L_;
#pragma unroll
  for (int j = 0; j < 8; ++j) prow[t + j * 256] = (__bf16)(e[j] * inv);
}

// ---------------------------------------------------------------------------
// Kernel 4: out_f32[b,q,d] = P[b,q,:] @ V[b,:,d]   (both bf16)
//   P tile is a pure copy -> async; V tile needs transpose -> VGPR staging.
// ---------------------------------------------------------------------------
__global__ __launch_bounds__(128) void pv_gemm_kernel(
    const __bf16* __restrict__ P, const __bf16* __restrict__ Vb,
    float* __restrict__ Out) {
  __shared__ __align__(64) __bf16 As[128][32];  // P tile [q][k]
  __shared__ __align__(64) __bf16 Bs[128][32];  // V tile transposed [d][k]

  const int t = threadIdx.x;
  const int lane = t & 31;
  const int wave = t >> 5;
  const int waveM = wave & 1, waveN = wave >> 1;
  const int b = blockIdx.z;
  const int q0 = blockIdx.y * 128;
  const int d0 = blockIdx.x * 128;

  const __bf16* Pbase = P + (size_t)b * L_ * L_;
  const __bf16* Vbase = Vb + (size_t)b * L_ * D_;

  v8f acc[4][4];
#pragma unroll
  for (int i = 0; i < 4; ++i)
#pragma unroll
    for (int j = 0; j < 4; ++j) acc[i][j] = (v8f){0, 0, 0, 0, 0, 0, 0, 0};

  const int bKr = t >> 2;        // 0..31
  const int bDc = (t & 3) * 32;  // 0,32,64,96

  for (int k0 = 0; k0 < L_; k0 += 32) {
    // A tile: async copy, one 64B row per thread
    const char* pg = (const char*)&Pbase[(size_t)(q0 + t) * L_ + k0];
    const char* pl = (const char*)&As[t][0];
#pragma unroll
    for (int j = 0; j < 4; ++j) async_b128(pg + j * 16, pl + j * 16);

    // B tile transposed: Bs[d][k]
    const __bf16* vg = &Vbase[(size_t)(k0 + bKr) * D_ + d0 + bDc];
#pragma unroll
    for (int j = 0; j < 32; ++j) Bs[bDc + j][bKr] = vg[j];
    if (k0 + 32 < L_)
      __builtin_prefetch(&Vbase[(size_t)(k0 + 32 + bKr) * D_ + d0 + bDc], 0, 1);

    async_wait0();
    __syncthreads();

    v16bf af[4], bf[4];
#pragma unroll
    for (int i = 0; i < 4; ++i)
      af[i] = *(const v16bf*)&As[waveM * 64 + i * 16 + (lane & 15)][(lane >> 4) * 16];
#pragma unroll
    for (int j = 0; j < 4; ++j)
      bf[j] = *(const v16bf*)&Bs[waveN * 64 + j * 16 + (lane & 15)][(lane >> 4) * 16];

#pragma unroll
    for (int i = 0; i < 4; ++i)
#pragma unroll
      for (int j = 0; j < 4; ++j)
        acc[i][j] = __builtin_amdgcn_wmma_f32_16x16x32_bf16(
            false, af[i], false, bf[j], (short)0, acc[i][j], false, false);
    __syncthreads();
  }

#pragma unroll
  for (int i = 0; i < 4; ++i) {
    const int rBase = q0 + waveM * 64 + i * 16 + ((lane >> 4) ? 8 : 0);
#pragma unroll
    for (int j = 0; j < 4; ++j) {
      const int c = d0 + waveN * 64 + j * 16 + (lane & 15);
#pragma unroll
      for (int e = 0; e < 8; ++e)
        Out[(size_t)(b * L_ + rBase + e) * D_ + c] = acc[i][j][e];
    }
  }
}

// ---------------------------------------------------------------------------
extern "C" void kernel_launch(void* const* d_in, const int* in_sizes, int n_in,
                              void* d_out, int out_size, void* d_ws,
                              size_t ws_size, hipStream_t stream) {
  (void)in_sizes; (void)n_in; (void)out_size; (void)ws_size;
  const float* input_q = (const float*)d_in[0];
  const float* input_k = (const float*)d_in[1];
  const float* k_mask  = (const float*)d_in[2];
  const float* Wq      = (const float*)d_in[3];
  const float* bq      = (const float*)d_in[4];
  const float* Wk      = (const float*)d_in[5];
  const float* bk      = (const float*)d_in[6];

  char* ws = (char*)d_ws;
  __bf16* Vb    = (__bf16*)(ws + OFF_V);
  __bf16* Qb    = (__bf16*)(ws + OFF_Q);
  __bf16* Kb    = (__bf16*)(ws + OFF_K);
  float*  mvals = (float*)(ws + OFF_M);
  float*  S     = (float*)(ws + OFF_S);
  __bf16* P     = (__bf16*)(ws + OFF_P);  // aliases Q/K after scores done
  float*  out   = (float*)d_out;

  prep_kernel<<<(M_TOTAL * D_ + 255) / 256, 256, 0, stream>>>(input_k, k_mask,
                                                              Vb, mvals);
  dim3 gProj(H_ / 128, M_TOTAL / 128);  // (8, 128)
  proj_gemm_kernel<<<gProj, 128, 0, stream>>>(input_q, Wq, bq, Qb, M_TOTAL, H_, D_);
  proj_gemm_kernel<<<gProj, 128, 0, stream>>>(input_k, Wk, bk, Kb, M_TOTAL, H_, D_);

  dim3 gQK(L_ / 128, L_ / 128, B_);  // (16, 16, 8)
  qk_scores_kernel<<<gQK, 128, 0, stream>>>(Qb, Kb, mvals, S);

  softmax_kernel<<<M_TOTAL, 256, 0, stream>>>(S, P);

  dim3 gPV(D_ / 128, L_ / 128, B_);  // (8, 16, 8)
  pv_gemm_kernel<<<gPV, 128, 0, stream>>>(P, Vb, out);
}
